// UserLayer_6846177870358
// MI455X (gfx1250) — matrix operand order repaired and checked by reading, not compiled, not run on hardware
//
#include <hip/hip_runtime.h>

#define BS   32768
#define NI   128
#define HID  640          // 5 * 128
#define G3   1920         // 3 * HID
#define NO   256

typedef _Float16 half_t;
typedef __attribute__((ext_vector_type(8)))  _Float16 v8h;
typedef __attribute__((ext_vector_type(16))) _Float16 v16h;
typedef __attribute__((ext_vector_type(8)))  float    v8f;

static __device__ __forceinline__ v16h cat8(v8h lo, v8h hi) {
  return __builtin_shufflevector(lo, hi, 0,1,2,3,4,5,6,7,8,9,10,11,12,13,14,15);
}

static __device__ __forceinline__ float sigmoidf_(float x) {
  return 1.0f / (1.0f + __expf(-x));
}

// ---------------------------------------------------------------------------
// Swizzle W[R x 640] f32 (row-major) into B-fragment-native f16 layout:
//   Wsw[((ks*RT + rt)*32 + kk)*16 + e] = (f16) W[rt*16 + kk%16][ks*32 + 16*(e/8) + 8*(kk/16) + e%8]
// Each (ks, rt) "blob" is 1 KB contiguous; lane kk's fragment = 32 contiguous B.
// ---------------------------------------------------------------------------
__global__ void k_convertW(const float* __restrict__ W, half_t* __restrict__ Wsw,
                           int RT) {
  int tid  = blockIdx.x * blockDim.x + threadIdx.x;  // exactly RT*16*640 threads
  int e    = tid & 15;
  int kk   = (tid >> 4) & 31;
  int blob = tid >> 9;
  int rt   = blob % RT;
  int ks   = blob / RT;
  Wsw[tid] = (half_t)W[(size_t)(rt * 16 + (kk & 15)) * HID + (ks * 32 + ((e & 8) << 1) + 8 * (kk >> 4) + (e & 7))];
}

// ---------------------------------------------------------------------------
__global__ void k_gather(const int* __restrict__ feat,
                         const float* __restrict__ e0, const float* __restrict__ e1,
                         const float* __restrict__ e2, const float* __restrict__ e3,
                         const float* __restrict__ e4,
                         float* __restrict__ h) {
  int tid = blockIdx.x * blockDim.x + threadIdx.x;   // BS*HID threads
  int b   = tid / HID;
  int j   = tid - b * HID;
  int nd  = j >> 7;
  int c   = j & 127;
  int row = feat[b * 5 + nd];
  const float* tab = (nd == 0) ? e0 : (nd == 1) ? e1 : (nd == 2) ? e2
                   : (nd == 3) ? e3 : e4;
  h[tid] = tab[(size_t)row * NI + c];
}

// ---------------------------------------------------------------------------
__global__ void k_msg(const float* __restrict__ h,
                      half_t* __restrict__ X, half_t* __restrict__ Hh) {
  int tid = blockIdx.x * blockDim.x + threadIdx.x;   // BS*128 threads
  int b = tid >> 7;
  int c = tid & 127;
  size_t base = (size_t)b * HID + c;
  float v0 = h[base +   0], v1 = h[base + 128], v2 = h[base + 256];
  float v3 = h[base + 384], v4 = h[base + 512];
  float s  = v0 + v1 + v2 + v3 + v4;
  X[base +   0] = (half_t)(s - v0);  Hh[base +   0] = (half_t)v0;
  X[base + 128] = (half_t)(s - v1);  Hh[base + 128] = (half_t)v1;
  X[base + 256] = (half_t)(s - v2);  Hh[base + 256] = (half_t)v2;
  X[base + 384] = (half_t)(s - v3);  Hh[base + 384] = (half_t)v3;
  X[base + 512] = (half_t)(s - v4);  Hh[base + 512] = (half_t)v4;
}

// ---------------------------------------------------------------------------
// Async-copy one K-stage of B fragments (6 x 1KB blobs) into LDS.
// 768 b64 transfers, exactly 3 per thread -> every wave issues 3 ASYNCcnt ops.
// ---------------------------------------------------------------------------
static __device__ __forceinline__ void stage_copy(
    const half_t* __restrict__ WihSw, const half_t* __restrict__ WhhSw,
    half_t* ldsbuf, int ks, int js, int tid) {
  #pragma unroll
  for (int it = 0; it < 3; ++it) {
    int o    = it * 256 + tid;          // 0..767
    int q    = o >> 7;                  // gate block 0..5
    int boff = (o & 127) * 8;           // byte offset within 1KB blob
    const half_t* base = (q < 3) ? WihSw : WhhSw;
    int qq = (q < 3) ? q : q - 3;
    const char* src =
        (const char*)(base + ((size_t)(ks * 120 + qq * 40 + js) << 9)) + boff;
    unsigned dst = (unsigned)(size_t)ldsbuf + q * 1024 + boff;
    asm volatile("global_load_async_to_lds_b64 %0, %1, off"
                 :: "v"(dst), "v"(src) : "memory");
  }
}

// ---------------------------------------------------------------------------
// Fused GRU round. Block = 8 waves, one 16-column gate slab (js), waves own
// 64 batch rows each. B fragments double-buffered in LDS via async-to-LDS;
// 24 v_wmma_f32_16x16x32_f16 per K-step per wave; gate math in-register.
// ---------------------------------------------------------------------------
__global__ __launch_bounds__(256) void k_gru(
    const half_t* __restrict__ X, const half_t* __restrict__ Hh,
    const half_t* __restrict__ WihSw, const half_t* __restrict__ WhhSw,
    const float* __restrict__ b_ih, const float* __restrict__ b_hh,
    float* __restrict__ h) {
  __shared__ __align__(16) half_t ldsB[2][6 * 512];  // 2 x 6KB

  const int tid  = threadIdx.x;
  const int lane = tid & 31;
  const int wid  = tid >> 5;
  const int mb   = blockIdx.x / 40;        // 0..63
  const int js   = blockIdx.x - mb * 40;   // 0..39 : 16-column gate slab
  const int mbase = (mb * 8 + wid) * 64;   // 64 batch rows per wave
  const int j0   = js * 16;
  const int g    = lane >> 4;
  const int m    = lane & 15;

  v8f acc[6][4];
  #pragma unroll
  for (int q = 0; q < 6; ++q)
    #pragma unroll
    for (int t = 0; t < 4; ++t) acc[q][t] = (v8f)(0.0f);

  const half_t* xbase = X  + (size_t)(mbase + m) * HID + 8 * g;
  const half_t* hbase = Hh + (size_t)(mbase + m) * HID + 8 * g;

  // prologue: stage 0 into buffer 0
  stage_copy(WihSw, WhhSw, &ldsB[0][0], 0, js, tid);

  for (int ks = 0; ks < HID / 32; ++ks) {
    const int p = ks & 1;
    if (ks < HID / 32 - 1) {
      stage_copy(WihSw, WhhSw, &ldsB[1 - p][0], ks + 1, js, tid);
      asm volatile("s_wait_asynccnt 0x3" ::: "memory");  // stage ks landed
    } else {
      asm volatile("s_wait_asynccnt 0x0" ::: "memory");
    }
    __syncthreads();

    // A fragments: 4 m-subtiles x {X, H}
    v16h af[4], hf[4];
    #pragma unroll
    for (int ms = 0; ms < 4; ++ms) {
      const half_t* xr = xbase + (size_t)(ms * 16) * HID + ks * 32;
      const half_t* hr = hbase + (size_t)(ms * 16) * HID + ks * 32;
      af[ms] = cat8(*(const v8h*)xr, *(const v8h*)(xr + 16));
      hf[ms] = cat8(*(const v8h*)hr, *(const v8h*)(hr + 16));
    }

    #pragma unroll
    for (int q = 0; q < 6; ++q) {
      const v8h* fp = (const v8h*)(&ldsB[p][q * 512 + lane * 16]);
      v16h bf = cat8(fp[0], fp[1]);
      #pragma unroll
      for (int ms = 0; ms < 4; ++ms) {
        acc[q][ms] = __builtin_amdgcn_wmma_f32_16x16x32_f16(
            false, (q < 3) ? af[ms] : hf[ms], false, bf,
            (short)0, acc[q][ms], false, false);
      }
    }
    __syncthreads();
  }

  // Gate math + in-place h update (C/D layout: n = lane&15, m = d + 8g)
  const int col = j0 + m;
  const float bir = b_ih[col], biz = b_ih[HID + col], binn = b_ih[2 * HID + col];
  const float bhr = b_hh[col], bhz = b_hh[HID + col], bhn  = b_hh[2 * HID + col];
  #pragma unroll
  for (int ms = 0; ms < 4; ++ms) {
    #pragma unroll
    for (int d = 0; d < 8; ++d) {
      const int row = mbase + ms * 16 + d + 8 * g;
      const size_t idx = (size_t)row * HID + col;
      const float hold = h[idx];
      const float r  = sigmoidf_(acc[0][ms][d] + bir + acc[3][ms][d] + bhr);
      const float z  = sigmoidf_(acc[1][ms][d] + biz + acc[4][ms][d] + bhz);
      const float nn = tanhf(acc[2][ms][d] + binn + r * (acc[5][ms][d] + bhn));
      h[idx] = (1.0f - z) * nn + z * hold;
    }
  }
}

// ---------------------------------------------------------------------------
__global__ void k_h2f16(const float* __restrict__ h, half_t* __restrict__ Hh) {
  int tid = blockIdx.x * blockDim.x + threadIdx.x;
  Hh[tid] = (half_t)h[tid];
}

// ---------------------------------------------------------------------------
// Final FC: out[32768 x 256] = Hh @ W_fc^T + b_fc  (WMMA, f32 accumulate)
// ---------------------------------------------------------------------------
__global__ __launch_bounds__(256) void k_fc(
    const half_t* __restrict__ Hh, const half_t* __restrict__ WfcSw,
    const float* __restrict__ b_fc, float* __restrict__ out) {
  const int lane = threadIdx.x & 31;
  const int wid  = threadIdx.x >> 5;
  const int gw   = blockIdx.x * 8 + wid;   // 0 .. 8191
  const int mt   = gw >> 2;
  const int js   = gw & 3;
  const int m0   = mt * 16;
  const int j0   = js * 64;
  const int g    = lane >> 4;
  const int m    = lane & 15;

  v8f acc[4];
  #pragma unroll
  for (int t = 0; t < 4; ++t) acc[t] = (v8f)(0.0f);

  const int RT = NO / 16;                  // 16 row-tiles in W_fc
  const half_t* hrow = Hh + (size_t)(m0 + m) * HID + 8 * g;

  for (int ks = 0; ks < HID / 32; ++ks) {
    v16h a = cat8(*(const v8h*)(hrow + ks * 32), *(const v8h*)(hrow + ks * 32 + 16));
    #pragma unroll
    for (int t = 0; t < 4; ++t) {
      const half_t* bp = WfcSw + (((size_t)ks * RT + js * 4 + t) * 32 + lane) * 16;
      v16h bf = cat8(*(const v8h*)bp, *(const v8h*)(bp + 8));
      acc[t] = __builtin_amdgcn_wmma_f32_16x16x32_f16(
          false, a, false, bf, (short)0, acc[t], false, false);
    }
  }

  #pragma unroll
  for (int t = 0; t < 4; ++t) {
    const int col = j0 + t * 16 + m;
    const float bias = b_fc[col];
    #pragma unroll
    for (int d = 0; d < 8; ++d) {
      const int row = m0 + d + 8 * g;
      out[(size_t)row * NO + col] = acc[t][d] + bias;
    }
  }
}

// ---------------------------------------------------------------------------
extern "C" void kernel_launch(void* const* d_in, const int* in_sizes, int n_in,
                              void* d_out, int out_size, void* d_ws, size_t ws_size,
                              hipStream_t stream) {
  (void)in_sizes; (void)n_in; (void)out_size; (void)ws_size;
  const int*   feat   = (const int*)  d_in[0];
  const float* emb_id = (const float*)d_in[1];
  const float* emb_ag = (const float*)d_in[2];
  const float* emb_ge = (const float*)d_in[3];
  const float* emb_oc = (const float*)d_in[4];
  const float* emb_zp = (const float*)d_in[5];
  const float* W_ih   = (const float*)d_in[6];
  const float* W_hh   = (const float*)d_in[7];
  const float* b_ih   = (const float*)d_in[8];
  const float* b_hh   = (const float*)d_in[9];
  const float* W_fc   = (const float*)d_in[10];
  const float* b_fc   = (const float*)d_in[11];

  char*  ws  = (char*)d_ws;
  size_t off = 0;
  auto alloc = [&](size_t bytes) -> void* {
    void* p = ws + off;
    off += (bytes + 255) & ~(size_t)255;
    return p;
  };
  float*  h     = (float*) alloc((size_t)BS * HID * 4);
  half_t* X     = (half_t*)alloc((size_t)BS * HID * 2);
  half_t* Hh    = (half_t*)alloc((size_t)BS * HID * 2);
  half_t* WihSw = (half_t*)alloc((size_t)G3 * HID * 2);
  half_t* WhhSw = (half_t*)alloc((size_t)G3 * HID * 2);
  half_t* WfcSw = (half_t*)alloc((size_t)NO * HID * 2);

  k_convertW<<<120 * 40, 256, 0, stream>>>(W_ih, WihSw, 120);
  k_convertW<<<120 * 40, 256, 0, stream>>>(W_hh, WhhSw, 120);
  k_convertW<<<16 * 40,  256, 0, stream>>>(W_fc, WfcSw, 16);

  k_gather<<<(BS * HID) / 256, 256, 0, stream>>>(feat, emb_id, emb_ag, emb_ge,
                                                 emb_oc, emb_zp, h);
  for (int r = 0; r < 2; ++r) {
    k_msg<<<(BS * 128) / 256, 256, 0, stream>>>(h, X, Hh);
    k_gru<<<2560, 256, 0, stream>>>(X, Hh, WihSw, WhhSw, b_ih, b_hh, h);
  }
  k_h2f16<<<(BS * HID) / 256, 256, 0, stream>>>(h, Hh);
  k_fc<<<1024, 256, 0, stream>>>(Hh, WfcSw, b_fc, (float*)d_out);
}
